// GAT_64836826301091
// MI455X (gfx1250) — compile-verified, matrix operand order checked by reading, not run on hardware
//
#include <hip/hip_runtime.h>
#include <stdint.h>

// ---------------- problem constants (match reference) ----------------
constexpr int kN   = 20000;   // nodes
constexpr int kFIN = 64;      // input features
constexpr int kH   = 4;       // heads
constexpr int kC   = 128;     // channels per head
constexpr int kHC  = 512;     // H*C
constexpr int kE   = 320000;  // edges (without self loops)
constexpr int kET  = kE + kN; // edges + self loops
constexpr int kFTS = 128;     // hidden linear width

// ---------------- types for WMMA ----------------
typedef __bf16 bf16x16 __attribute__((ext_vector_type(16)));
typedef float  f32x8   __attribute__((ext_vector_type(8)));

union FragBF { bf16x16 v; uint4 q[2]; };

// async global->LDS copy availability (gfx1250 path; fallback = load+ds_store)
#if defined(__has_builtin)
#if __has_builtin(__builtin_amdgcn_global_load_async_to_lds_b128)
#define USE_ASYNC_LDS 1
#endif
#endif
#ifndef USE_ASYNC_LDS
#define USE_ASYNC_LDS 0
#endif

// builtin expects: (int4 __device__* /*addrspace(1)*/, int4 __shared__* /*addrspace(3)*/, imm, imm)
typedef int v4i __attribute__((vector_size(16)));
typedef __attribute__((address_space(1))) v4i* glob_v4i_p;
typedef __attribute__((address_space(3))) v4i* lds_v4i_p;

__device__ __forceinline__ void wait_asynccnt0() {
#if defined(__has_builtin) && __has_builtin(__builtin_amdgcn_s_wait_asynccnt)
  __builtin_amdgcn_s_wait_asynccnt(0);
#else
  asm volatile("s_wait_asynccnt 0x0" ::: "memory");
#endif
}

// fp32 -> bf16 round-to-nearest-even
__device__ __forceinline__ unsigned short f2bf(float f) {
  unsigned u = __float_as_uint(f);
  u += 0x7FFFu + ((u >> 16) & 1u);
  return (unsigned short)(u >> 16);
}

// monotonic uint key for float atomic-max
__device__ __forceinline__ unsigned fkey(float f) {
  unsigned u = __float_as_uint(f);
  return (u & 0x80000000u) ? ~u : (u | 0x80000000u);
}
__device__ __forceinline__ float funkey(unsigned k) {
  unsigned u = (k & 0x80000000u) ? (k & 0x7FFFFFFFu) : ~k;
  return __uint_as_float(u);
}

// ---------------- elementwise helpers ----------------
__global__ void k_f2bf(const float* __restrict__ in, unsigned short* __restrict__ out, int n) {
  int i = blockIdx.x * blockDim.x + threadIdx.x;
  if (i < n) out[i] = f2bf(in[i]);
}

__global__ void k_fill_u32(unsigned* __restrict__ p, unsigned v, int n) {
  int i = blockIdx.x * blockDim.x + threadIdx.x;
  if (i < n) p[i] = v;
}

// mean of edge weights, single block (cheap: 1.28 MB)
__global__ void k_mean(const float* __restrict__ ew, float* __restrict__ meanw, int n) {
  __shared__ float sh[256];
  float s = 0.f;
  for (int i = threadIdx.x; i < n; i += 256) s += ew[i];
  sh[threadIdx.x] = s;
  __syncthreads();
  for (int o = 128; o; o >>= 1) {
    if (threadIdx.x < (unsigned)o) sh[threadIdx.x] += sh[threadIdx.x + o];
    __syncthreads();
  }
  if (threadIdx.x == 0) *meanw = sh[0] / (float)n;
}

// ce[h] = sum_c We[h*C+c] * ae[h*C+c]   (grid = H, block = 128)
__global__ void k_ce(const float* __restrict__ We, const float* __restrict__ ae,
                     float* __restrict__ ce) {
  __shared__ float sh[128];
  int h = blockIdx.x, c = threadIdx.x;
  sh[c] = We[h * kC + c] * ae[h * kC + c];
  __syncthreads();
  for (int o = 64; o; o >>= 1) {
    if (c < o) sh[c] += sh[c + o];
    __syncthreads();
  }
  if (c == 0) ce[h] = sh[0];
}

// ---------------- WMMA GEMM: C[M,Nout] = A[M,K](bf16) x B[Nout,K](bf16)^T ----
// Launch contract: blockDim.x = (Nout/64)*32, i.e. every wave owns exactly
// 4 valid 16x16 N-tiles (a 16x64 strip) -> no guards, EXEC all-1s for WMMA.
// Block owns 16 M-rows; the A K-chunk (16x128) is staged in LDS once per
// block (async global->LDS), then one A fragment feeds 4 back-to-back WMMAs.
// grid.x = M/16.
__global__ void k_gemm_bf16(const unsigned short* __restrict__ A,
                            const unsigned short* __restrict__ B,
                            float* __restrict__ C,
                            const float* __restrict__ bias,
                            int Nout, int K, int relu) {
  __shared__ __align__(16) unsigned short shA[16 * 128];  // 4 KB A chunk
  const int tid  = threadIdx.x;
  const int nthr = blockDim.x;
  const int wave = tid >> 5;
  const int lane = tid & 31;
  const int mt   = blockIdx.x;

  // 16-bit fragment lane layout (ISA 7.12.2): row = lane&15,
  // kbase = (lane>>4)*8; elems 0..7 <- K=kbase+0..7, elems 8..15 <- K=kbase+16..23
  const int row   = lane & 15;
  const int kbase = (lane >> 4) * 8;

  const unsigned short* brow[4];
#pragma unroll
  for (int j = 0; j < 4; ++j)
    brow[j] = B + (size_t)((wave * 4 + j) * 16 + row) * K + kbase;

  f32x8 acc[4];
#pragma unroll
  for (int j = 0; j < 4; ++j) acc[j] = {0.f, 0.f, 0.f, 0.f, 0.f, 0.f, 0.f, 0.f};

  for (int k0 = 0; k0 < K; k0 += 128) {
    const int chunkK = (K - k0 < 128) ? (K - k0) : 128;
    const int qpr    = chunkK >> 3;   // 16-byte loads per A row
    __syncthreads();                  // previous chunk fully consumed
    for (int t = tid; t < 16 * qpr; t += nthr) {
      const int r = t / qpr;
      const int c = (t % qpr) * 8;
      const unsigned short* gp = A + (size_t)(mt * 16 + r) * K + k0 + c;
      unsigned short* lp = &shA[r * 128 + c];
#if USE_ASYNC_LDS
      // gfx1250 async copy: global -> LDS, tracked by ASYNCcnt
      __builtin_amdgcn_global_load_async_to_lds_b128(
          (glob_v4i_p)(uintptr_t)gp, (lds_v4i_p)(unsigned)(uintptr_t)lp, 0, 0);
#else
      *(uint4*)lp = *(const uint4*)gp;
#endif
    }
#if USE_ASYNC_LDS
    wait_asynccnt0();
#endif
    __syncthreads();                  // chunk visible to all waves

    for (int kk = 0; kk < chunkK; kk += 32) {
      FragBF a;
      a.q[0] = *(const uint4*)&shA[row * 128 + kk + kbase];       // ds_load_b128
      a.q[1] = *(const uint4*)&shA[row * 128 + kk + kbase + 16];
#pragma unroll
      for (int j = 0; j < 4; ++j) {
        const unsigned short* bp = brow[j] + k0 + kk;
        FragBF b;
        b.q[0] = *(const uint4*)bp;
        b.q[1] = *(const uint4*)(bp + 16);
        __builtin_prefetch(bp + 32, 0, 3);  // global_prefetch_b8
        acc[j] = __builtin_amdgcn_wmma_f32_16x16x32_bf16(
            /*neg_a=*/false, a.v, /*neg_b=*/false, b.v,
            /*c_mod=*/(short)0, acc[j], /*reuse_a=*/false, /*reuse_b=*/false);
      }
    }
  }

  // C/D layout: VGPR r -> M = r + 8*(lane>>4), N = lane&15
  const int mbase = mt * 16 + ((lane >> 4) << 3);
#pragma unroll
  for (int j = 0; j < 4; ++j) {
    const int col = (wave * 4 + j) * 16 + (lane & 15);
    const float bv = bias ? bias[col] : 0.f;
#pragma unroll
    for (int r = 0; r < 8; ++r) {
      float v = acc[j][r] + bv;
      if (relu) v = fmaxf(v, 0.f);
      C[(size_t)(mbase + r) * Nout + col] = v;
    }
  }
}

// ---------------- alpha_src / alpha_dst: one wave32 per (n,h) ---------------
__global__ void k_alphas(const float* __restrict__ Hm,
                         const float* __restrict__ a_s, const float* __restrict__ a_d,
                         float* __restrict__ asrc, float* __restrict__ adst) {
  int wid  = (blockIdx.x * blockDim.x + threadIdx.x) >> 5;
  int lane = threadIdx.x & 31;
  if (wid >= kN * kH) return;
  int n = wid >> 2, h = wid & 3;
  const float* hr = Hm + (size_t)n * kHC + h * kC;
  float s = 0.f, d = 0.f;
  for (int c = lane; c < kC; c += 32) {
    float hv = hr[c];
    s += hv * a_s[h * kC + c];
    d += hv * a_d[h * kC + c];
  }
  for (int o = 16; o; o >>= 1) {
    s += __shfl_down(s, o, 32);
    d += __shfl_down(d, o, 32);
  }
  if (lane == 0) { asrc[wid] = s; adst[wid] = d; }
}

// ---------------- edge pass 1: leaky-relu logits + segment max ---------------
__global__ void k_edge_logits(const long long* __restrict__ ei, const float* __restrict__ ew,
                              const float* __restrict__ meanw,
                              const float* __restrict__ asrc, const float* __restrict__ adst,
                              const float* __restrict__ ce,
                              float* __restrict__ alphaE, unsigned* __restrict__ mx) {
  int t = blockIdx.x * blockDim.x + threadIdx.x;
  if (t >= kET * kH) return;
  int e = t >> 2, h = t & 3;
  long long s, d; float w;
  if (e < kE) { s = ei[e]; d = ei[kE + e]; w = ew[e]; }
  else        { s = d = e - kE;            w = *meanw; }
  float v = asrc[s * kH + h] + adst[d * kH + h] + w * ce[h];
  v = (v > 0.f) ? v : 0.2f * v;  // leaky_relu(0.2)
  alphaE[t] = v;
  atomicMax(&mx[d * kH + h], fkey(v));
}

// ---------------- edge pass 2: exp + segment sum ----------------------------
__global__ void k_edge_exp(const long long* __restrict__ ei,
                           const unsigned* __restrict__ mx,
                           float* __restrict__ alphaE, float* __restrict__ sm) {
  int t = blockIdx.x * blockDim.x + threadIdx.x;
  if (t >= kET * kH) return;
  int e = t >> 2, h = t & 3;
  long long d = (e < kE) ? ei[kE + e] : (long long)(e - kE);
  float p = __expf(alphaE[t] - funkey(mx[d * kH + h]));
  alphaE[t] = p;
  atomicAdd(&sm[d * kH + h], p);
}

// ---------------- edge pass 3: acc[dst] += h[src] * alpha -------------------
// block = 512 threads (one per channel), grid = kET (one block per edge)
__global__ void k_edge_aggr(const long long* __restrict__ ei,
                            const float* __restrict__ alphaE, const float* __restrict__ sm,
                            const float* __restrict__ Hm, float* __restrict__ acc) {
  int e = blockIdx.x, c = threadIdx.x;
  long long s, d;
  if (e < kE) { s = ei[e]; d = ei[kE + e]; }
  else        { s = d = e - kE; }
  int h = c >> 7;  // c / 128
  float alpha = alphaE[(size_t)e * kH + h] / (sm[d * kH + h] + 1e-16f);
  atomicAdd(&acc[(size_t)d * kHC + c], Hm[(size_t)s * kHC + c] * alpha);
}

// ---------------- bias + relu + bf16 convert --------------------------------
__global__ void k_bias_relu_bf16(const float* __restrict__ acc, const float* __restrict__ b,
                                 unsigned short* __restrict__ outb, int total) {
  int i = blockIdx.x * blockDim.x + threadIdx.x;
  if (i >= total) return;
  int col = i & (kHC - 1);
  outb[i] = f2bf(fmaxf(acc[i] + b[col], 0.f));
}

// ---------------- final head dot: one wave32 per node -----------------------
__global__ void k_head_final(const float* __restrict__ Hh, const float* __restrict__ L1W,
                             const float* __restrict__ L1b, float* __restrict__ out) {
  int wid  = (blockIdx.x * blockDim.x + threadIdx.x) >> 5;
  int lane = threadIdx.x & 31;
  if (wid >= kN) return;
  const float* hr = Hh + (size_t)wid * kFTS;
  float s = 0.f;
  for (int c = lane; c < kFTS; c += 32) s += hr[c] * L1W[c];
  for (int o = 16; o; o >>= 1) s += __shfl_down(s, o, 32);
  if (lane == 0) out[wid] = s + L1b[0];
}

// ==========================================================================
extern "C" void kernel_launch(void* const* d_in, const int* in_sizes, int n_in,
                              void* d_out, int out_size, void* d_ws, size_t ws_size,
                              hipStream_t stream) {
  (void)in_sizes; (void)n_in; (void)out_size; (void)ws_size;
  const float*     x   = (const float*)d_in[0];
  const long long* ei  = (const long long*)d_in[1];
  const float*     ew  = (const float*)d_in[2];
  const float*     W0  = (const float*)d_in[3];
  const float*     as0 = (const float*)d_in[4];
  const float*     ad0 = (const float*)d_in[5];
  const float*     We0 = (const float*)d_in[6];
  const float*     ae0 = (const float*)d_in[7];
  const float*     b0  = (const float*)d_in[8];
  const float*     W1  = (const float*)d_in[9];
  const float*     as1 = (const float*)d_in[10];
  const float*     ad1 = (const float*)d_in[11];
  const float*     We1 = (const float*)d_in[12];
  const float*     ae1 = (const float*)d_in[13];
  const float*     b1  = (const float*)d_in[14];
  const float*     L0W = (const float*)d_in[15];
  const float*     L0b = (const float*)d_in[16];
  const float*     L1W = (const float*)d_in[17];
  const float*     L1b = (const float*)d_in[18];
  float* out = (float*)d_out;

  // ---- carve workspace ----
  char* ws = (char*)d_ws;
  size_t off = 0;
  auto alloc = [&](size_t bytes) -> char* {
    char* p = ws + off;
    off = (off + bytes + 255) & ~(size_t)255;
    return p;
  };
  unsigned short* bX   = (unsigned short*)alloc((size_t)kN * kFIN * 2);
  unsigned short* bW0  = (unsigned short*)alloc((size_t)kHC * kFIN * 2);
  unsigned short* bW1  = (unsigned short*)alloc((size_t)kHC * kHC * 2);
  unsigned short* bLW  = (unsigned short*)alloc((size_t)kFTS * kHC * 2);
  unsigned short* bAct = (unsigned short*)alloc((size_t)kN * kHC * 2);
  float*    bufH   = (float*)alloc((size_t)kN * kHC * 4);
  float*    bufAcc = (float*)alloc((size_t)kN * kHC * 4);
  float*    asrc   = (float*)alloc((size_t)kN * kH * 4);
  float*    adst   = (float*)alloc((size_t)kN * kH * 4);
  unsigned* mx     = (unsigned*)alloc((size_t)kN * kH * 4);
  float*    sm     = (float*)alloc((size_t)kN * kH * 4);
  float*    alphaE = (float*)alloc((size_t)kET * kH * 4);
  float*    ceBuf  = (float*)alloc(8 * 4);  // ce0[0..3], ce1[4..7]
  float*    meanw  = (float*)alloc(4);

  auto cdiv = [](int a, int b) { return (a + b - 1) / b; };

  // ---- precompute: bf16 weights/input, mean(ew), ce per layer ----
  k_f2bf<<<cdiv(kN * kFIN, 256), 256, 0, stream>>>(x, bX, kN * kFIN);
  k_f2bf<<<cdiv(kHC * kFIN, 256), 256, 0, stream>>>(W0, bW0, kHC * kFIN);
  k_f2bf<<<cdiv(kHC * kHC, 256), 256, 0, stream>>>(W1, bW1, kHC * kHC);
  k_f2bf<<<cdiv(kFTS * kHC, 256), 256, 0, stream>>>(L0W, bLW, kFTS * kHC);
  k_mean<<<1, 256, 0, stream>>>(ew, meanw, kE);
  k_ce<<<kH, 128, 0, stream>>>(We0, ae0, ceBuf);
  k_ce<<<kH, 128, 0, stream>>>(We1, ae1, ceBuf + 4);

  const int nNH  = kN * kH;
  const int nEH  = kET * kH;
  const int nAcc = kN * kHC;
  const int gEH  = cdiv(nEH, 256);
  const int gNH8 = cdiv(nNH, 8);  // 8 waves/block in k_alphas

  // ================= GAT layer as a host-side routine =================
  auto gat_layer = [&](const unsigned short* Ab, int K, const unsigned short* Wb,
                       const float* a_s, const float* a_d, const float* ce,
                       const float* bias) {
    // h = A @ W^T  (fp32 result in bufH); Nout = 512 -> 8 waves of 16x64
    k_gemm_bf16<<<kN / 16, (kHC / 64) * 32, 0, stream>>>(Ab, Wb, bufH, nullptr, kHC, K, 0);
    // per-node attention scalars
    k_alphas<<<gNH8, 256, 0, stream>>>(bufH, a_s, a_d, asrc, adst);
    // softmax over incoming edges
    k_fill_u32<<<cdiv(nNH, 256), 256, 0, stream>>>(mx, 0u, nNH);
    k_fill_u32<<<cdiv(nNH, 256), 256, 0, stream>>>((unsigned*)sm, 0u, nNH);
    k_edge_logits<<<gEH, 256, 0, stream>>>(ei, ew, meanw, asrc, adst, ce, alphaE, mx);
    k_edge_exp<<<gEH, 256, 0, stream>>>(ei, mx, alphaE, sm);
    // weighted aggregation
    k_fill_u32<<<cdiv(nAcc, 256), 256, 0, stream>>>((unsigned*)bufAcc, 0u, nAcc);
    k_edge_aggr<<<kET, 512, 0, stream>>>(ei, alphaE, sm, bufH, bufAcc);
    // bias + relu -> bf16 activation for the next GEMM
    k_bias_relu_bf16<<<cdiv(nAcc, 256), 256, 0, stream>>>(bufAcc, bias, bAct, nAcc);
  };

  gat_layer(bX, kFIN, bW0, as0, ad0, ceBuf, b0);       // layer 0: x -> bAct
  gat_layer(bAct, kHC, bW1, as1, ad1, ceBuf + 4, b1);  // layer 1: bAct -> bAct

  // ---- head: relu(bAct @ L0W^T + L0b) fused, then dot with L1W ----
  // Nout = 128 -> 2 waves of 16x64 per block
  k_gemm_bf16<<<kN / 16, (kFTS / 64) * 32, 0, stream>>>(bAct, bLW, bufH, L0b, kFTS, kHC, 1);
  k_head_final<<<cdiv(kN, 8), 256, 0, stream>>>(bufH, L1W, L1b, out);
}
